// RnnLM_13855564497003
// MI455X (gfx1250) — matrix operand order repaired and checked by reading, not compiled, not run on hardware
//
#include <hip/hip_runtime.h>
#include <math.h>

// ---------------------------------------------------------------------------
// RNN LM for MI455X (gfx1250, wave32, WMMA + TDM).
// Phase 1: convert weights f32->bf16 into workspace, embed ids[:,0].
// Phase 2: sequential recurrence (128 t x 4 layers), each layer = 2 kernels,
//          64x64 tiles, v_wmma_f32_16x16x32_bf16, f32 accumulation, tanh.
// Phase 3: one big logits GEMM (8192x1024)x(1024x32000) -> f32 (B,T,V).
//          Operand staging via Tensor Data Mover (tensor_load_to_lds) with
//          double-buffered LDS and TENSORcnt-tracked overlap; wte (65 MB
//          bf16) stays resident in the 192 MB L2.
// ---------------------------------------------------------------------------

typedef __attribute__((ext_vector_type(16))) __bf16 v16bf;
typedef __attribute__((ext_vector_type(8)))  float  v8f;
typedef unsigned int u32x4 __attribute__((ext_vector_type(4)));
typedef unsigned int tdm_g0 __attribute__((ext_vector_type(4)));
typedef int          tdm_g1 __attribute__((ext_vector_type(8)));
typedef int          tdm_g23 __attribute__((ext_vector_type(4)));

#define RNN_DIM   1024
#define RNN_NL    4
#define RNN_VOCAB 32000
#define RNN_B     64
#define RNN_T     128
#define KSTEP     32
#define LDSK      40   // padded LDS row stride in bf16 elems (80 B, 16B-aligned)

union Frag { u32x4 u[2]; v16bf v; };

// A-matrix 16x32 bf16 fragment (ISA 7.12.2): lane=(m, hi); VGPR v0..3 hold
// k = hi*8 + {0..7}, v4..7 hold k = 16 + hi*8 + {0..7}, packed pairwise.
__device__ inline v16bf lds_a_frag(const __bf16* p, int m_sub, int lane) {
  int m = m_sub + (lane & 15);
  int hi = lane >> 4;
  Frag f;
  f.u[0] = *(const u32x4*)(p + m * LDSK + hi * 8);
  f.u[1] = *(const u32x4*)(p + m * LDSK + 16 + hi * 8);
  return f.v;
}

// B-matrix 32x16 bf16 fragment: lane=(n, hi); v0..3 hold k = hi*8 + {0..7},
// v4..7 hold k = 16 + hi*8 + {0..7} (same order as the A fragment).
__device__ inline v16bf lds_b_frag(const __bf16* p, int n_sub, int lane) {
  int n = n_sub + (lane & 15);
  int hi = lane >> 4;
  Frag f;
  f.u[0] = *(const u32x4*)(p + n * LDSK + hi * 8);
  f.u[1] = *(const u32x4*)(p + n * LDSK + 16 + hi * 8);
  return f.v;
}

// Issue one TDM 2D tile load: 32 (k, contiguous) x 128 (rows) bf16 from a
// row-major [rows x 1024] tensor into LDS with 16B padding after every 64B
// (row stride 80B == LDSK*2). D# per ISA 8.3/8.4: count=1, type=2 (image),
// data_size=1 (2B), pad_interval=3 (16 DW), pad_amount=3 (4 DW).
// This toolchain exposes the 6-arg builtin (g0, g1, g2, g3, extra, cpol).
__device__ inline void tdm_load_tile(const __bf16* gbase, unsigned rows,
                                     unsigned lds_byte, int row0, int k0) {
  unsigned long long ga = (unsigned long long)gbase +
      ((unsigned long long)row0 * RNN_DIM + (unsigned)k0) * 2ull;
  tdm_g0 g0 = { 1u,                                   // count=1
                lds_byte,                             // lds_addr
                (unsigned)ga,                         // global_addr[31:0]
                (unsigned)(ga >> 32) | (2u << 30) };  // ga[56:32] | type=2
  tdm_g1 g1 = {
      (int)((1u << 16) | (1u << 20) | (3u << 22) | (3u << 25)),
      (int)((RNN_DIM & 0xFFFFu) << 16),               // tensor_dim0 lo16
      (int)((RNN_DIM >> 16) | ((rows & 0xFFFFu) << 16)),  // dim0 hi | dim1 lo
      (int)((rows >> 16) | ((unsigned)KSTEP << 16)),  // dim1 hi | tile_dim0
      128,                                            // tile_dim1 (tile_dim2=0)
      RNN_DIM,                                        // tensor_dim0_stride lo32
      0, 0 };
  tdm_g23 gz4 = { 0, 0, 0, 0 };
  tdm_g1 gz8 = { 0, 0, 0, 0, 0, 0, 0, 0 };
  __builtin_amdgcn_tensor_load_to_lds(g0, g1, gz4, gz4, gz8, 0);
}

// ------------------------- setup kernels -----------------------------------

__global__ void k_cvt_bf16(const float* __restrict__ in,
                           __bf16* __restrict__ out, int n) {
  int i = blockIdx.x * blockDim.x + threadIdx.x;
  if (i < n) out[i] = (__bf16)in[i];
}

__global__ void k_zero_u32(unsigned int* __restrict__ p, int n) {
  int i = blockIdx.x * blockDim.x + threadIdx.x;
  if (i < n) p[i] = 0u;
}

__global__ void k_embed(const int* __restrict__ ids,
                        const float* __restrict__ wte,
                        __bf16* __restrict__ xbuf) {
  int i = blockIdx.x * blockDim.x + threadIdx.x;   // RNN_B*RNN_DIM threads
  int b = i >> 10;
  int d = i & (RNN_DIM - 1);
  long row = (long)ids[b * RNN_T];                 // ids[:,0]
  xbuf[i] = (__bf16)wte[row * RNN_DIM + d];
}

// ------------------------- recurrence kernels ------------------------------
// h_t = tanh(h @ Whh.T + bhh + x @ Wxh.T + bxh); M=64, N=1024, K=1024 (x2)
__global__ __launch_bounds__(128) void k_rnn_h(
    const __bf16* __restrict__ h_in, const __bf16* __restrict__ x_in,
    const __bf16* __restrict__ Whh, const __bf16* __restrict__ Wxh,
    const float* __restrict__ bhh, const float* __restrict__ bxh,
    __bf16* __restrict__ h_out) {
  __shared__ __bf16 Alds[64 * LDSK];
  __shared__ __bf16 Blds[64 * LDSK];
  const int tid = threadIdx.x;
  const int lane = tid & 31, wave = tid >> 5;
  const int n0 = blockIdx.x * 64;
  const int m_sub = wave * 16;
  const int r0 = tid >> 2, j0 = tid & 3;   // chunk ownership (branch-free)

  v8f acc[4];
  v8f vz = {};
#pragma unroll
  for (int j = 0; j < 4; ++j) acc[j] = vz;

  for (int pass = 0; pass < 2; ++pass) {
    const __bf16* A = pass ? x_in : h_in;
    const __bf16* W = pass ? Wxh : Whh;
    for (int k0 = 0; k0 < RNN_DIM; k0 += KSTEP) {
      // 64x32 bf16 tiles: 256 16-byte chunks each, thread owns {tid, tid+128}.
      *(u32x4*)(Alds + r0 * LDSK + j0 * 8) =
          *(const u32x4*)(A + r0 * RNN_DIM + k0 + j0 * 8);
      *(u32x4*)(Alds + (r0 + 32) * LDSK + j0 * 8) =
          *(const u32x4*)(A + (r0 + 32) * RNN_DIM + k0 + j0 * 8);
      *(u32x4*)(Blds + r0 * LDSK + j0 * 8) =
          *(const u32x4*)(W + (long)(n0 + r0) * RNN_DIM + k0 + j0 * 8);
      *(u32x4*)(Blds + (r0 + 32) * LDSK + j0 * 8) =
          *(const u32x4*)(W + (long)(n0 + r0 + 32) * RNN_DIM + k0 + j0 * 8);
      __syncthreads();
      v16bf a = lds_a_frag(Alds, m_sub, lane);
#pragma unroll
      for (int j = 0; j < 4; ++j) {
        v16bf b = lds_b_frag(Blds, j * 16, lane);
        acc[j] = __builtin_amdgcn_wmma_f32_16x16x32_bf16(
            false, a, false, b, (short)0, acc[j], false, false);
      }
      __syncthreads();
    }
  }

  const int hi = lane >> 4, nl = lane & 15;
#pragma unroll
  for (int j = 0; j < 4; ++j) {
    int n = n0 + j * 16 + nl;
    float bias = bhh[n] + bxh[n];
#pragma unroll
    for (int r = 0; r < 8; ++r) {
      int m = m_sub + hi * 8 + r;
      h_out[m * RNN_DIM + n] = (__bf16)tanhf(acc[j][r] + bias);
    }
  }
}

// x_t = tanh(h_t @ Who.T + bho); M=64, N=1024, K=1024
__global__ __launch_bounds__(128) void k_rnn_x(
    const __bf16* __restrict__ h_in, const __bf16* __restrict__ Who,
    const float* __restrict__ bho,
    __bf16* __restrict__ x_out, __bf16* __restrict__ x_slot) {
  __shared__ __bf16 Alds[64 * LDSK];
  __shared__ __bf16 Blds[64 * LDSK];
  const int tid = threadIdx.x;
  const int lane = tid & 31, wave = tid >> 5;
  const int n0 = blockIdx.x * 64;
  const int m_sub = wave * 16;
  const int r0 = tid >> 2, j0 = tid & 3;

  v8f acc[4];
  v8f vz = {};
#pragma unroll
  for (int j = 0; j < 4; ++j) acc[j] = vz;

  for (int k0 = 0; k0 < RNN_DIM; k0 += KSTEP) {
    *(u32x4*)(Alds + r0 * LDSK + j0 * 8) =
        *(const u32x4*)(h_in + r0 * RNN_DIM + k0 + j0 * 8);
    *(u32x4*)(Alds + (r0 + 32) * LDSK + j0 * 8) =
        *(const u32x4*)(h_in + (r0 + 32) * RNN_DIM + k0 + j0 * 8);
    *(u32x4*)(Blds + r0 * LDSK + j0 * 8) =
        *(const u32x4*)(Who + (long)(n0 + r0) * RNN_DIM + k0 + j0 * 8);
    *(u32x4*)(Blds + (r0 + 32) * LDSK + j0 * 8) =
        *(const u32x4*)(Who + (long)(n0 + r0 + 32) * RNN_DIM + k0 + j0 * 8);
    __syncthreads();
    v16bf a = lds_a_frag(Alds, m_sub, lane);
#pragma unroll
    for (int j = 0; j < 4; ++j) {
      v16bf b = lds_b_frag(Blds, j * 16, lane);
      acc[j] = __builtin_amdgcn_wmma_f32_16x16x32_bf16(
          false, a, false, b, (short)0, acc[j], false, false);
    }
    __syncthreads();
  }

  const int hi = lane >> 4, nl = lane & 15;
#pragma unroll
  for (int j = 0; j < 4; ++j) {
    int n = n0 + j * 16 + nl;
    float bias = bho[n];
#pragma unroll
    for (int r = 0; r < 8; ++r) {
      int m = m_sub + hi * 8 + r;
      __bf16 v = (__bf16)tanhf(acc[j][r] + bias);
      x_out[m * RNN_DIM + n] = v;
      if (x_slot) x_slot[m * RNN_DIM + n] = v;
    }
  }
}

// ------------------------- logits GEMM (TDM pipelined) ---------------------
// (M=8192 rows = t*64+b) x (N=32000) over K=1024. 128x128 tile per block,
// 8 waves, each wave 32x64 = 2x4 wmma accumulators. Operands staged by the
// Tensor Data Mover into double-buffered padded LDS; wave 0 issues DMA for
// tile k+1 while all waves run WMMA on tile k. Output f32 in (B,T,V).
#define LOGITS_BUF_E (128 * LDSK)     // one buffer, bf16 elems (10240 B)

__global__ __launch_bounds__(256) void k_logits(
    const __bf16* __restrict__ xall, const __bf16* __restrict__ wteb,
    float* __restrict__ out) {
  __shared__ __bf16 Alds[2 * LOGITS_BUF_E];
  __shared__ __bf16 Blds[2 * LOGITS_BUF_E];
  const int tid = threadIdx.x;
  const int lane = tid & 31, wave = tid >> 5;
  const int n0 = blockIdx.x * 128;   // 250 tiles
  const int m0 = blockIdx.y * 128;   // 64 tiles
  const int wm = wave & 3;           // 0..3 -> m base wm*32
  const int wn = wave >> 2;          // 0..1 -> n base wn*64

  const unsigned a_lds = (unsigned)(unsigned long long)(void*)Alds;
  const unsigned b_lds = (unsigned)(unsigned long long)(void*)Blds;
  const unsigned buf_bytes = LOGITS_BUF_E * 2u;

  v8f acc[2][4];
  v8f vz = {};
#pragma unroll
  for (int i = 0; i < 2; ++i)
#pragma unroll
    for (int j = 0; j < 4; ++j) acc[i][j] = vz;

  // Prologue: DMA tile 0 into buffer 0.
  if (wave == 0) {
    tdm_load_tile(xall, RNN_T * RNN_B, a_lds, m0, 0);
    tdm_load_tile(wteb, RNN_VOCAB, b_lds, n0, 0);
  }

  const int NK = RNN_DIM / KSTEP;  // 32
  for (int kk = 0; kk < NK; ++kk) {
    const int cur = kk & 1;
    if (wave == 0) {
      if (kk + 1 < NK) {  // DMA next tile into the other buffer
        tdm_load_tile(xall, RNN_T * RNN_B, a_lds + (cur ^ 1) * buf_bytes,
                      m0, (kk + 1) * KSTEP);
        tdm_load_tile(wteb, RNN_VOCAB, b_lds + (cur ^ 1) * buf_bytes,
                      n0, (kk + 1) * KSTEP);
        __builtin_amdgcn_s_wait_tensorcnt(2);  // in-order: current tile done
      } else {
        __builtin_amdgcn_s_wait_tensorcnt(0);
      }
    }
    __syncthreads();  // current buffer visible to all waves

    const __bf16* Ab = Alds + cur * LOGITS_BUF_E;
    const __bf16* Bb = Blds + cur * LOGITS_BUF_E;
    v16bf a0 = lds_a_frag(Ab, wm * 32, lane);
    v16bf a1 = lds_a_frag(Ab, wm * 32 + 16, lane);
#pragma unroll
    for (int j = 0; j < 4; ++j) {
      v16bf b = lds_b_frag(Bb, wn * 64 + j * 16, lane);
      acc[0][j] = __builtin_amdgcn_wmma_f32_16x16x32_bf16(
          false, a0, false, b, (short)0, acc[0][j], false, false);
      acc[1][j] = __builtin_amdgcn_wmma_f32_16x16x32_bf16(
          false, a1, false, b, (short)0, acc[1][j], false, false);
    }
    __syncthreads();  // all waves done with this buffer before reuse
  }

  const int hi = lane >> 4, nl = lane & 15;
#pragma unroll
  for (int i = 0; i < 2; ++i) {
#pragma unroll
    for (int j = 0; j < 4; ++j) {
      int ng = n0 + wn * 64 + j * 16 + nl;
#pragma unroll
      for (int r = 0; r < 8; ++r) {
        int mg = m0 + wm * 32 + i * 16 + hi * 8 + r;
        int t = mg >> 6;          // row = t*64 + b
        int b = mg & 63;
        out[(long)b * ((long)RNN_T * RNN_VOCAB) + (long)t * RNN_VOCAB + ng] =
            acc[i][j][r];
      }
    }
  }
}

// ------------------------- host launcher -----------------------------------

extern "C" void kernel_launch(void* const* d_in, const int* in_sizes, int n_in,
                              void* d_out, int out_size, void* d_ws,
                              size_t ws_size, hipStream_t stream) {
  (void)in_sizes; (void)n_in; (void)out_size; (void)ws_size;

  const int*   ids   = (const int*)d_in[0];
  const float* wte   = (const float*)d_in[1];
  const float* wxh_w = (const float*)d_in[2];
  const float* wxh_b = (const float*)d_in[3];
  const float* whh_w = (const float*)d_in[4];
  const float* whh_b = (const float*)d_in[5];
  const float* who_w = (const float*)d_in[6];
  const float* who_b = (const float*)d_in[7];
  float* out = (float*)d_out;

  const size_t WTE_E = (size_t)RNN_VOCAB * RNN_DIM;           // 32,768,000
  const size_t W_E   = (size_t)RNN_NL * RNN_DIM * RNN_DIM;    // 4,194,304
  const size_t XA_E  = (size_t)RNN_T * RNN_B * RNN_DIM;       // 8,388,608
  const size_t XB_E  = (size_t)RNN_B * RNN_DIM;               // 65,536
  const size_t HB_E  = (size_t)2 * RNN_NL * RNN_B * RNN_DIM;  // 524,288

  __bf16* ws   = (__bf16*)d_ws;
  __bf16* wteb = ws;                 size_t o = WTE_E;
  __bf16* wxhb = ws + o;             o += W_E;
  __bf16* whhb = ws + o;             o += W_E;
  __bf16* whob = ws + o;             o += W_E;
  __bf16* xall = ws + o;             o += XA_E;
  __bf16* xbuf = ws + o;             o += XB_E;
  __bf16* hbuf = ws + o;             o += HB_E;

  // Phase 1: conversions + init (re-done every call: deterministic).
  k_cvt_bf16<<<(int)(WTE_E / 256), 256, 0, stream>>>(wte, wteb, (int)WTE_E);
  k_cvt_bf16<<<(int)(W_E / 256), 256, 0, stream>>>(wxh_w, wxhb, (int)W_E);
  k_cvt_bf16<<<(int)(W_E / 256), 256, 0, stream>>>(whh_w, whhb, (int)W_E);
  k_cvt_bf16<<<(int)(W_E / 256), 256, 0, stream>>>(who_w, whob, (int)W_E);
  k_zero_u32<<<(int)(HB_E / 2 / 256), 256, 0, stream>>>((unsigned int*)hbuf,
                                                        (int)(HB_E / 2));
  k_embed<<<(int)(XB_E / 256), 256, 0, stream>>>(ids, wte, xbuf);

  // Phase 2: sequential recurrence, double-buffered hidden state by t parity.
  const size_t HS = (size_t)RNN_B * RNN_DIM;     // one layer's h, elems
  const size_t WS = (size_t)RNN_DIM * RNN_DIM;   // one layer's W, elems
  for (int t = 0; t < RNN_T; ++t) {
    int pr = t & 1;
    for (int l = 0; l < RNN_NL; ++l) {
      __bf16* hin  = hbuf + ((size_t)pr * RNN_NL + l) * HS;
      __bf16* hout = hbuf + ((size_t)(1 - pr) * RNN_NL + l) * HS;
      k_rnn_h<<<16, 128, 0, stream>>>(hin, xbuf, whhb + (size_t)l * WS,
                                      wxhb + (size_t)l * WS,
                                      whh_b + (size_t)l * RNN_DIM,
                                      wxh_b + (size_t)l * RNN_DIM, hout);
      __bf16* slot = (l == RNN_NL - 1) ? (xall + (size_t)t * HS)
                                       : (__bf16*)nullptr;
      k_rnn_x<<<16, 128, 0, stream>>>(hout, whob + (size_t)l * WS,
                                      who_b + (size_t)l * RNN_DIM, xbuf, slot);
    }
  }

  // Phase 3: one big logits GEMM over all timesteps (TDM-pipelined).
  dim3 grid(RNN_VOCAB / 128, (RNN_T * RNN_B) / 128);
  k_logits<<<grid, 256, 0, stream>>>(xall, wteb, out);
}
